// Attention_25159918420763
// MI455X (gfx1250) — matrix operand order repaired and checked by reading, not compiled, not run on hardware
//
#include <hip/hip_runtime.h>
#include <hip/hip_bf16.h>

typedef __attribute__((ext_vector_type(16))) _Float16 v16h;
typedef __attribute__((ext_vector_type(8)))  float    v8f;
typedef __attribute__((ext_vector_type(4)))  float    v4f;

#define BS 64
#define QKD 128

// ---------------------------------------------------------------------------
// block-wide sum for 256-thread blocks (8 wave32s). Result valid in thread 0.
// (cold-path helper; not used in the streaming kernel)
// ---------------------------------------------------------------------------
__device__ __forceinline__ float blockReduce256(float v, float* lds) {
    #pragma unroll
    for (int o = 16; o > 0; o >>= 1) v += __shfl_down(v, o, 32);
    int lane = threadIdx.x & 31, w = threadIdx.x >> 5;
    __syncthreads();                  // protect lds reuse across calls
    if (lane == 0) lds[w] = v;
    __syncthreads();
    if (threadIdx.x < 8) {
        v = lds[threadIdx.x];
        #pragma unroll
        for (int o = 4; o > 0; o >>= 1) v += __shfl_down(v, o, 32);
    }
    return v;
}

// ---------------------------------------------------------------------------
// Phase A: fused single-pass streaming reduction over X[b, C, HW]:
//   chan_part[b, c, chunk, wave]  = per-wave sum_{hw} X        (per c in split)
//   sq_part[b, split, hw]         = sum_{c in split} X^2       (per hw in chunk)
// grid = (nch, csplit, BS), block = 256, each thread owns 4 consecutive hw.
// Hot loop: 1 b128 NT load + 7 VALU + 5 shfl per wave per c. No LDS/barriers.
// ---------------------------------------------------------------------------
__global__ void reduce_main(const float* __restrict__ X,
                            float* __restrict__ chan_part,
                            float* __restrict__ sq_part,
                            int C, int HW, int nch, int cper)
{
    const int chunk = blockIdx.x, split = blockIdx.y, b = blockIdx.z;
    const int tid = threadIdx.x;
    const int lane = tid & 31, wid = tid >> 5;
    const int hw0 = chunk * 1024 + tid * 4;
    const bool vec = ((HW & 3) == 0);
    const float* Xb = X + (size_t)b * C * HW;
    const int c0 = split * cper;
    float a0 = 0.f, a1 = 0.f, a2 = 0.f, a3 = 0.f;

    for (int cc = 0; cc < cper; ++cc) {
        const int c = c0 + cc;
        const float* row = Xb + (size_t)c * HW + hw0;
        float x0 = 0.f, x1 = 0.f, x2 = 0.f, x3 = 0.f;
        if (vec) {
            if (hw0 < HW) {           // HW%4==0 -> whole float4 in-range
                v4f v4 = __builtin_nontemporal_load((const v4f*)row);
                x0 = v4.x; x1 = v4.y; x2 = v4.z; x3 = v4.w;
            }
        } else {
            if (hw0 + 0 < HW) x0 = __builtin_nontemporal_load(row + 0);
            if (hw0 + 1 < HW) x1 = __builtin_nontemporal_load(row + 1);
            if (hw0 + 2 < HW) x2 = __builtin_nontemporal_load(row + 2);
            if (hw0 + 3 < HW) x3 = __builtin_nontemporal_load(row + 3);
        }
        a0 += x0 * x0; a1 += x1 * x1; a2 += x2 * x2; a3 += x3 * x3;
        float ps = (x0 + x1) + (x2 + x3);
        #pragma unroll
        for (int o = 16; o > 0; o >>= 1) ps += __shfl_down(ps, o, 32);
        if (lane == 0)
            chan_part[(((size_t)b * C + c) * nch + chunk) * 8 + wid] = ps;
    }
    float* sp = sq_part + ((size_t)b * gridDim.y + split) * HW;
    if (vec) {
        if (hw0 < HW) {
            v4f o; o.x = a0; o.y = a1; o.z = a2; o.w = a3;
            *(v4f*)(sp + hw0) = o;
        }
    } else {
        if (hw0 + 0 < HW) sp[hw0 + 0] = a0;
        if (hw0 + 1 < HW) sp[hw0 + 1] = a1;
        if (hw0 + 2 < HW) sp[hw0 + 2] = a2;
        if (hw0 + 3 < HW) sp[hw0 + 3] = a3;
    }
}

// chan_mean[b,c] = (1/HW) * sum over nch*8 per-wave partials (deterministic)
__global__ void combine_chan(const float* __restrict__ part, float* __restrict__ mean,
                             int np, int total, float invHW)
{
    int idx = blockIdx.x * 256 + threadIdx.x;
    if (idx >= total) return;
    float s = 0.f;
    for (int k = 0; k < np; ++k) s += part[(size_t)idx * np + k];
    mean[idx] = s * invHW;
}

// sqmap[b,hw] = (1/C) * sum_split sq_part
__global__ void combine_sq(const float* __restrict__ part, float* __restrict__ sqmap,
                           int csplit, int HW, int total, float invC)
{
    int idx = blockIdx.x * 256 + threadIdx.x;
    if (idx >= total) return;
    int b = idx / HW, hw = idx - b * HW;
    float s = 0.f;
    for (int k = 0; k < csplit; ++k) s += part[((size_t)b * csplit + k) * HW + hw];
    sqmap[idx] = s * invC;
}

// ---------------------------------------------------------------------------
// WMMA GEMM: Out[M,N] = A[M,K] * B[K,N]  (f32 data, f16 hi/lo split, f32 acc)
// grid = (N/16, M/16), block = 32 (one wave per 16x16 tile). K % 32 == 0.
// ---------------------------------------------------------------------------
__global__ void gemm_wmma(const float* __restrict__ A, int lda,
                          const float* __restrict__ Bm, int ldb,
                          float* __restrict__ Out, int ldo, int K)
{
    const int nt = blockIdx.x, mt = blockIdx.y;
    const int l = threadIdx.x;
    const int m  = l & 15;              // A-tile row for this lane
    const int hb = (l >> 4) * 8;        // A K-half-group base
    const int n  = l & 15;              // B-tile col
    const int kb2 = (l >> 4) * 16;      // B K base
    v8f acc = {};
    const float* Arow = A + (size_t)(mt * 16 + m) * lda;
    const float* Bcol = Bm + nt * 16 + n;

    for (int k0 = 0; k0 < K; k0 += 32) {
        v16h ah, al, bh, bl;
        #pragma unroll
        for (int j = 0; j < 16; ++j) {
            // A 16x32 f16 layout: K = (l>>4)*16 + j
            int ka = k0 + kb2 + j;
            float av = Arow[ka];
            _Float16 h = (_Float16)av;
            ah[j] = h; al[j] = (_Float16)(av - (float)h);
            // B 32x16 f16 layout: K = (l>>4)*16 + j
            float bv = Bcol[(size_t)(k0 + kb2 + j) * ldb];
            _Float16 hbv = (_Float16)bv;
            bh[j] = hbv; bl[j] = (_Float16)(bv - (float)hbv);
        }
        acc = __builtin_amdgcn_wmma_f32_16x16x32_f16(false, ah, false, bh, (short)0, acc, false, false);
        acc = __builtin_amdgcn_wmma_f32_16x16x32_f16(false, ah, false, bl, (short)0, acc, false, false);
        acc = __builtin_amdgcn_wmma_f32_16x16x32_f16(false, al, false, bh, (short)0, acc, false, false);
    }
    // C/D layout: VGPR r -> row = mt*16 + r + 8*(l>>4), col = nt*16 + (l&15)
    const int rbase = mt * 16 + (l >> 4) * 8;
    const int col = nt * 16 + (l & 15);
    #pragma unroll
    for (int r = 0; r < 8; ++r)
        Out[(size_t)(rbase + r) * ldo + col] = acc[r];
}

// ---------------------------------------------------------------------------
// Batch-norm (training: batch stats, biased variance) over `rows` per column,
// optional ReLU. Bias before BN cancels -> omitted.
// ---------------------------------------------------------------------------
__global__ void bn_kernel(const float* __restrict__ raw, int rows, int cols,
                          const float* __restrict__ g, const float* __restrict__ beta,
                          float* __restrict__ out, int ostride, int relu)
{
    int col = blockIdx.x * blockDim.x + threadIdx.x;
    if (col >= cols) return;
    float s = 0.f;
    for (int r = 0; r < rows; ++r) s += raw[(size_t)r * cols + col];
    float m = s / rows;
    float v = 0.f;
    for (int r = 0; r < rows; ++r) { float d = raw[(size_t)r * cols + col] - m; v += d * d; }
    float inv = rsqrtf(v / rows + 1e-5f);
    float gg = g[col], bb = beta[col];
    for (int r = 0; r < rows; ++r) {
        float y = (raw[(size_t)r * cols + col] - m) * inv * gg + bb;
        if (relu) y = fmaxf(y, 0.f);
        out[(size_t)r * ostride + col] = y;
    }
}

// logit[b,t,s] = sum_q bkey[(b*4+s)*512 + t*128 + q] * query[b*512 + t*128 + q]
// atts = softmax over s. One thread per batch element.
__global__ void logit_softmax(const float* __restrict__ bkey,
                              const float* __restrict__ query,
                              float* __restrict__ atts)
{
    int b = threadIdx.x;
    if (b >= BS) return;
    float lg[4][4];
    for (int t = 0; t < 4; ++t)
        for (int s = 0; s < 4; ++s) {
            const float* kk = bkey + (size_t)(b * 4 + s) * 512 + t * QKD;
            const float* qq = query + (size_t)b * 512 + t * QKD;
            float d = 0.f;
            for (int q = 0; q < QKD; ++q) d += kk[q] * qq[q];
            lg[t][s] = d;
        }
    for (int t = 0; t < 4; ++t) {
        float mx = lg[t][0];
        for (int s = 1; s < 4; ++s) mx = fmaxf(mx, lg[t][s]);
        float e[4], sum = 0.f;
        for (int s = 0; s < 4; ++s) { e[s] = __expf(lg[t][s] - mx); sum += e[s]; }
        for (int s = 0; s < 4; ++s) atts[b * 16 + t * 4 + s] = e[s] / sum;
    }
}

// ---------------------------------------------------------------------------
// Pooled attention-map MSE: mean((vs-vt)^2) with l2-normalized maps reduces to
// (ss/ns^2 - 2*dot/(ns*nt) + tt/nt^2)/N. Adaptive pool = kxk box avg (down)
// or k-replication (up) for the sizes {56,28,14,7}.
// grid = (16 = i*4+j, 64 = b), block = 256.
// ---------------------------------------------------------------------------
struct Maps { const float* s[4]; const float* t[4]; };
__constant__ __device__ const int kDIM[4] = {56, 28, 14, 7};

__global__ void pooldot(Maps mp, float* __restrict__ diffbuf)
{
    const int i = blockIdx.x >> 2, j = blockIdx.x & 3, b = blockIdx.y;
    const int tN = kDIM[i], sN = kDIM[j];
    const int N = tN * tN;
    const float* sq_s = mp.s[j] + (size_t)b * sN * sN;
    const float* sq_t = mp.t[i] + (size_t)b * N;
    float ss = 0.f, dot = 0.f, tt = 0.f;
    if (sN >= tN) {
        const int k = sN / tN;
        const float invk2 = 1.f / (float)(k * k);
        for (int o = threadIdx.x; o < N; o += 256) {
            int oy = o / tN, ox = o - oy * tN;
            float sum = 0.f;
            for (int yy = 0; yy < k; ++yy) {
                const float* r = sq_s + (oy * k + yy) * sN + ox * k;
                for (int xx = 0; xx < k; ++xx) sum += r[xx];
            }
            float p = sum * invk2, tv = sq_t[o];
            ss += p * p; dot += p * tv; tt += tv * tv;
        }
    } else {
        const int k = tN / sN;
        for (int o = threadIdx.x; o < N; o += 256) {
            int oy = o / tN, ox = o - oy * tN;
            float p = sq_s[(oy / k) * sN + (ox / k)];
            float tv = sq_t[o];
            ss += p * p; dot += p * tv; tt += tv * tv;
        }
    }
    __shared__ float lds[8];
    ss  = blockReduce256(ss, lds);
    dot = blockReduce256(dot, lds);
    tt  = blockReduce256(tt, lds);
    if (threadIdx.x == 0) {
        float ns = fmaxf(sqrtf(ss), 1e-12f);
        float nt = fmaxf(sqrtf(tt), 1e-12f);
        float d = ss / (ns * ns) - 2.f * dot / (ns * nt) + tt / (nt * nt);
        diffbuf[(i * BS + b) * 4 + j] = d / (float)N;
    }
}

// loss[i] = mean_b sum_s diff[i,b,s] * atts[b,i,s]   (deterministic, 1 block)
__global__ void finalize(const float* __restrict__ diffbuf,
                         const float* __restrict__ atts,
                         float* __restrict__ out)
{
    __shared__ float lds[256];
    int tid = threadIdx.x;
    int i = tid >> 6, b = tid & 63;
    float v = 0.f;
    for (int s = 0; s < 4; ++s)
        v += diffbuf[(i * BS + b) * 4 + s] * atts[b * 16 + i * 4 + s];
    lds[i * 64 + b] = v;
    __syncthreads();
    if (tid < 4) {
        float s = 0.f;
        for (int bb = 0; bb < 64; ++bb) s += lds[tid * 64 + bb];
        out[tid] = s / 64.f;
    }
}

// ---------------------------------------------------------------------------
extern "C" void kernel_launch(void* const* d_in, const int* in_sizes, int n_in,
                              void* d_out, int out_size, void* d_ws, size_t ws_size,
                              hipStream_t stream)
{
    (void)in_sizes; (void)n_in; (void)out_size; (void)ws_size;
    static const int SCc[4] = {64, 128, 256, 512};
    static const int TCc[4] = {256, 512, 1024, 2048};
    static const int HWs[4] = {3136, 784, 196, 49};

    const float* gs[4]; const float* gt[4]; const float* Wk[4]; const float* Wq[4];
    for (int j = 0; j < 4; ++j) gs[j] = (const float*)d_in[j];
    for (int i = 0; i < 4; ++i) gt[i] = (const float*)d_in[4 + i];
    for (int j = 0; j < 4; ++j) Wk[j] = (const float*)d_in[8 + j];
    for (int i = 0; i < 4; ++i) Wq[i] = (const float*)d_in[12 + i];
    const float* gk    = (const float*)d_in[17];
    const float* betak = (const float*)d_in[18];
    const float* gq    = (const float*)d_in[20];
    const float* betaq = (const float*)d_in[21];
    const float* Wb    = (const float*)d_in[22];
    const float* gb    = (const float*)d_in[24];
    const float* betab = (const float*)d_in[25];

    // ---- workspace carve-up (floats) ----
    float* w = (float*)d_ws;
    size_t off = 0;
    auto alloc = [&](size_t n) { float* p = w + off; off += (n + 3) & ~(size_t)3; return p; };

    int nch_s[4], nch_t[4], csp_s[4], csp_t[4];
    float *cps[4], *cpt[4], *sps[4], *spt[4];
    float *mean_s[4], *mean_t[4], *sqmap_s[4], *sqmap_t[4];
    for (int j = 0; j < 4; ++j) {
        nch_s[j] = (HWs[j] + 1023) / 1024;  csp_s[j] = SCc[j] / 32;
        cps[j] = alloc((size_t)BS * SCc[j] * nch_s[j] * 8);
        sps[j] = alloc((size_t)BS * csp_s[j] * HWs[j]);
        mean_s[j]  = alloc((size_t)BS * SCc[j]);
        sqmap_s[j] = alloc((size_t)BS * HWs[j]);
    }
    for (int i = 0; i < 4; ++i) {
        nch_t[i] = (HWs[i] + 1023) / 1024;  csp_t[i] = TCc[i] / 32;
        cpt[i] = alloc((size_t)BS * TCc[i] * nch_t[i] * 8);
        spt[i] = alloc((size_t)BS * csp_t[i] * HWs[i]);
        mean_t[i]  = alloc((size_t)BS * TCc[i]);
        sqmap_t[i] = alloc((size_t)BS * HWs[i]);
    }
    float* keyraw[4]; float* qraw[4];
    for (int j = 0; j < 4; ++j) keyraw[j] = alloc((size_t)BS * QKD);
    for (int i = 0; i < 4; ++i) qraw[i]   = alloc((size_t)BS * QKD);
    float* key   = alloc((size_t)BS * 4 * QKD);   // [b, s, q]
    float* query = alloc((size_t)BS * 4 * QKD);   // [b, t, q]
    float* braw  = alloc((size_t)256 * 512);
    float* bkey  = alloc((size_t)256 * 512);      // [b*4+s, t*128+q]
    float* atts  = alloc((size_t)BS * 16);        // [b, t, s]
    float* diffb = alloc((size_t)4 * BS * 4);     // [i, b, j]

    // ---- Phase A: single-pass streaming reductions (HBM-bound, ~480 MB) ----
    for (int j = 0; j < 4; ++j)
        reduce_main<<<dim3(nch_s[j], csp_s[j], BS), 256, 0, stream>>>(
            gs[j], cps[j], sps[j], SCc[j], HWs[j], nch_s[j], 32);
    for (int i = 0; i < 4; ++i)
        reduce_main<<<dim3(nch_t[i], csp_t[i], BS), 256, 0, stream>>>(
            gt[i], cpt[i], spt[i], TCc[i], HWs[i], nch_t[i], 32);

    for (int j = 0; j < 4; ++j) {
        int tot = BS * SCc[j];
        combine_chan<<<(tot + 255) / 256, 256, 0, stream>>>(cps[j], mean_s[j], nch_s[j] * 8, tot, 1.f / HWs[j]);
        int tot2 = BS * HWs[j];
        combine_sq<<<(tot2 + 255) / 256, 256, 0, stream>>>(sps[j], sqmap_s[j], csp_s[j], HWs[j], tot2, 1.f / SCc[j]);
    }
    for (int i = 0; i < 4; ++i) {
        int tot = BS * TCc[i];
        combine_chan<<<(tot + 255) / 256, 256, 0, stream>>>(cpt[i], mean_t[i], nch_t[i] * 8, tot, 1.f / HWs[i]);
        int tot2 = BS * HWs[i];
        combine_sq<<<(tot2 + 255) / 256, 256, 0, stream>>>(spt[i], sqmap_t[i], csp_t[i], HWs[i], tot2, 1.f / TCc[i]);
    }

    // ---- Phase B: WMMA GEMMs + BN ----
    for (int j = 0; j < 4; ++j) {
        gemm_wmma<<<dim3(QKD / 16, BS / 16), 32, 0, stream>>>(
            mean_s[j], SCc[j], Wk[j], QKD, keyraw[j], QKD, SCc[j]);
        bn_kernel<<<1, 128, 0, stream>>>(keyraw[j], BS, QKD, gk + j * QKD, betak + j * QKD,
                                         key + j * QKD, 4 * QKD, 1);
    }
    for (int i = 0; i < 4; ++i) {
        gemm_wmma<<<dim3(QKD / 16, BS / 16), 32, 0, stream>>>(
            mean_t[i], TCc[i], Wq[i], QKD, qraw[i], QKD, TCc[i]);
        bn_kernel<<<1, 128, 0, stream>>>(qraw[i], BS, QKD, gq + i * QKD, betaq + i * QKD,
                                         query + i * QKD, 4 * QKD, 0);
    }
    // bilinear: [256,128] @ [128,512]
    gemm_wmma<<<dim3(512 / 16, 256 / 16), 32, 0, stream>>>(key, QKD, Wb, 512, braw, 512, QKD);
    bn_kernel<<<4, 128, 0, stream>>>(braw, 256, 512, gb, betab, bkey, 512, 0);

    // ---- Phase C: attention + pooled diffs + loss ----
    logit_softmax<<<1, 64, 0, stream>>>(bkey, query, atts);

    Maps mp;
    for (int j = 0; j < 4; ++j) mp.s[j] = sqmap_s[j];
    for (int i = 0; i < 4; ++i) mp.t[i] = sqmap_t[i];
    pooldot<<<dim3(16, 64), 256, 0, stream>>>(mp, diffb);

    finalize<<<1, 256, 0, stream>>>(diffb, atts, (float*)d_out);
}
